// WaveNet_27298812133957
// MI455X (gfx1250) — matrix operand order repaired and checked by reading, not compiled, not run on hardware
//
#include <hip/hip_runtime.h>

// ---------------------------------------------------------------------------
// WaveNet VQ-VAE forward for gfx1250 (MI455X), wave32 + v_wmma_f32_16x16x32_f16
// Weights pre-packed to f16 (fragment-friendly, zero-padded) in workspace.
// ---------------------------------------------------------------------------

typedef __attribute__((ext_vector_type(16))) _Float16 v16h;
typedef __attribute__((ext_vector_type(8)))  float    v8f;

#define T_LEN   16384
#define B_SZ    8
#define TT      64      // interior timesteps per block
#define SW      96      // stored rows (interior + 16 halo each side)
#define CW      80      // compute rows (interior + 8 each side)
#define SOFF    8       // compute-region offset inside stored region
#define DKP     160     // decoder padded K (129 -> 5 slabs of 32)
#define DNP     144     // decoder padded N (129 -> 9 tiles of 16)

// ---- f16 weight arena layout (element offsets) ----------------------------
constexpr size_t OFF_EF0  = 0;                          // [4][256][32] filt tap0
constexpr size_t OFF_EF1  = OFF_EF0 + 4 * 256 * 32;     // [4][256][32] filt tap1
constexpr size_t OFF_EG0  = OFF_EF1 + 4 * 256 * 32;     // [4][256][32] gate tap0
constexpr size_t OFF_EG1  = OFF_EG0 + 4 * 256 * 32;     // [4][256][32] gate tap1
constexpr size_t OFF_ES   = OFF_EG1 + 4 * 256 * 32;     // [4][256][256] skip
constexpr size_t OFF_ER   = OFF_ES  + 4 * 256 * 256;    // [4][32][256]  res
constexpr size_t OFF_EEND = OFF_ER  + 4 * 32 * 256;     // [256][256]    end
constexpr size_t OFF_PRE  = OFF_EEND + 256 * 256;       // [128][256]    pre
constexpr size_t OFF_EMB  = OFF_PRE + 128 * 256;        // [256][128]    codebook
constexpr size_t OFF_DST  = OFF_EMB + 256 * 128;        // [32][160]     dec start
constexpr size_t OFF_DF0  = OFF_DST + 32 * DKP;         // [4][144][32]
constexpr size_t OFF_DF1  = OFF_DF0 + 4 * DNP * 32;
constexpr size_t OFF_DG0  = OFF_DF1 + 4 * DNP * 32;
constexpr size_t OFF_DG1  = OFF_DG0 + 4 * DNP * 32;
constexpr size_t OFF_DSK  = OFF_DG1 + 4 * DNP * 32;     // [4][144][160]
constexpr size_t OFF_DR   = OFF_DSK + 4 * DNP * DKP;    // [4][32][160]
constexpr size_t W_TOTAL  = OFF_DR  + 4 * 32 * DKP;     // ~748K f16 elems

__device__ __forceinline__ v8f vzero() {
  v8f z;
#pragma unroll
  for (int i = 0; i < 8; ++i) z[i] = 0.0f;
  return z;
}

__device__ __forceinline__ v8f wmma16(v16h a, v16h b, v8f c) {
  // D = A(16x32 f16) * B(32x16 f16) + C(16x16 f32)
  return __builtin_amdgcn_wmma_f32_16x16x32_f16(false, a, false, b, (short)0, c,
                                                false, false);
}

// Branch-free native activations (v_exp_f32 / v_rcp_f32, no ocml slow paths).
__device__ __forceinline__ float fast_sigmoid(float x) {
  float e = __builtin_amdgcn_exp2f(-1.4426950408889634f * x);
  return __builtin_amdgcn_rcpf(1.0f + e);
}
__device__ __forceinline__ float fast_tanh(float x) {
  float ax = __builtin_fabsf(x);
  float e  = __builtin_amdgcn_exp2f(2.8853900817779268f * ax);  // 2^(2*log2e*|x|)
  float t  = 1.0f - 2.0f * __builtin_amdgcn_rcpf(1.0f + e);     // exp2->inf => t=1
  return x < 0.0f ? -t : t;                                     // v_cndmask
}

union AFrag { unsigned int u[8]; v16h h; };

// A fragment (16x32 f16): rows = M (time), cols = K (channels).
// Layout (ISA 7.12.2): lane m=L&15, half=L>>4; pair j<4 -> K=2j+8*half,
// pair j>=4 -> K=16+2(j-4)+8*half.
__device__ __forceinline__ v16h load_a_lds(const _Float16* base, int row0,
                                           int stride, int k0) {
  const int lane = threadIdx.x & 31;
  const int m = lane & 15, half = lane >> 4;
  const _Float16* rp = base + (row0 + m) * stride + k0 + 8 * half;
  AFrag a;
#pragma unroll
  for (int j = 0; j < 4; ++j) {
    a.u[j]     = *(const unsigned int*)(rp + 2 * j);
    a.u[4 + j] = *(const unsigned int*)(rp + 16 + 2 * j);
  }
  return a.h;
}

// A fragment from global f16 (row-major [row][stride]).
__device__ __forceinline__ v16h load_a_glb(const _Float16* base, long row0,
                                           int stride, int k0) {
  const int lane = threadIdx.x & 31;
  const int m = lane & 15, half = lane >> 4;
  const _Float16* rp = base + (row0 + m) * (long)stride + k0 + 8 * half;
  AFrag a;
#pragma unroll
  for (int j = 0; j < 4; ++j) {
    a.u[j]     = *(const unsigned int*)(rp + 2 * j);
    a.u[4 + j] = *(const unsigned int*)(rp + 16 + 2 * j);
  }
  return a.h;
}

// B fragment (32x16) from pre-packed f16 [n][k] matrix: lane n=L&15,
// half=L>>4; pair j<4 -> K=2j+8*half, pair j>=4 -> K=16+2(j-4)+8*half.
__device__ __forceinline__ v16h load_b16(const _Float16* base, int stride,
                                         int n0, int k0) {
  const int lane = threadIdx.x & 31;
  const int n = lane & 15, half = lane >> 4;
  const _Float16* rp = base + (size_t)(n0 + n) * stride + k0 + 8 * half;
  AFrag a;
#pragma unroll
  for (int j = 0; j < 4; ++j) {
    a.u[j]     = *(const unsigned int*)(rp + 2 * j);
    a.u[4 + j] = *(const unsigned int*)(rp + 16 + 2 * j);
  }
  return a.h;
}

// ---------------------------------------------------------------------------
// Pack: all weights -> f16 arena (de-interleaved taps, zero-padded dec mats)
// ---------------------------------------------------------------------------
__launch_bounds__(256)
__global__ void wn_pack_kernel(const float* __restrict__ efw,   // [4][256][32][2]
                               const float* __restrict__ egw,
                               const float* __restrict__ esw,   // [4][256][256]
                               const float* __restrict__ erw,   // [4][32][256]
                               const float* __restrict__ eew,   // [256][256]
                               const float* __restrict__ prw,   // [128][256]
                               const float* __restrict__ emb,   // [256][128]
                               const float* __restrict__ dstw,  // [32][129]
                               const float* __restrict__ dfw,   // [4][129][32][2]
                               const float* __restrict__ dgw,
                               const float* __restrict__ dsw,   // [4][129][129]
                               const float* __restrict__ drw,   // [4][32][129]
                               _Float16* __restrict__ w)
{
  const size_t tid = (size_t)blockIdx.x * blockDim.x + threadIdx.x;
  const size_t nth = (size_t)gridDim.x * blockDim.x;

  for (size_t i = tid; i < 4 * 256 * 32; i += nth) {
    w[OFF_EF0 + i] = (_Float16)efw[i * 2 + 0];
    w[OFF_EF1 + i] = (_Float16)efw[i * 2 + 1];
    w[OFF_EG0 + i] = (_Float16)egw[i * 2 + 0];
    w[OFF_EG1 + i] = (_Float16)egw[i * 2 + 1];
  }
  for (size_t i = tid; i < 4 * 256 * 256; i += nth) w[OFF_ES + i] = (_Float16)esw[i];
  for (size_t i = tid; i < 4 * 32 * 256;  i += nth) w[OFF_ER + i] = (_Float16)erw[i];
  for (size_t i = tid; i < 256 * 256;     i += nth) w[OFF_EEND + i] = (_Float16)eew[i];
  for (size_t i = tid; i < 128 * 256;     i += nth) w[OFF_PRE + i] = (_Float16)prw[i];
  for (size_t i = tid; i < 256 * 128;     i += nth) w[OFF_EMB + i] = (_Float16)emb[i];

  for (size_t i = tid; i < 32 * DKP; i += nth) {
    size_t o = i / DKP, k = i % DKP;
    w[OFF_DST + i] = (_Float16)((k < 129) ? dstw[o * 129 + k] : 0.0f);
  }
  for (size_t i = tid; i < 4 * DNP * 32; i += nth) {
    size_t l = i / (DNP * 32), r = i % (DNP * 32), o = r / 32, k = r % 32;
    float f0 = 0, f1 = 0, g0 = 0, g1 = 0;
    if (o < 129) {
      size_t s = ((l * 129 + o) * 32 + k) * 2;
      f0 = dfw[s]; f1 = dfw[s + 1]; g0 = dgw[s]; g1 = dgw[s + 1];
    }
    w[OFF_DF0 + i] = (_Float16)f0;
    w[OFF_DF1 + i] = (_Float16)f1;
    w[OFF_DG0 + i] = (_Float16)g0;
    w[OFF_DG1 + i] = (_Float16)g1;
  }
  for (size_t i = tid; i < 4 * DNP * DKP; i += nth) {
    size_t l = i / (DNP * DKP), r = i % (DNP * DKP), o = r / DKP, k = r % DKP;
    float v = (o < 129 && k < 129) ? dsw[(l * 129 + o) * 129 + k] : 0.0f;
    w[OFF_DSK + i] = (_Float16)v;
  }
  for (size_t i = tid; i < 4 * 32 * DKP; i += nth) {
    size_t l = i / (32 * DKP), r = i % (32 * DKP), o = r / DKP, k = r % DKP;
    float v = (k < 129) ? drw[(l * 32 + o) * 129 + k] : 0.0f;
    w[OFF_DR + i] = (_Float16)v;
  }
}

// ---------------------------------------------------------------------------
// Encoder: start conv + 4 WaveNet layers + end conv(tanh) + pre conv -> z f16
// ---------------------------------------------------------------------------
__launch_bounds__(256)
__global__ void wn_enc_kernel(const float* __restrict__ xin,
                              const float* __restrict__ start_w,  // [32] f32
                              const float* __restrict__ skip_b,   // [4][256] f32
                              const float* __restrict__ res_b,    // [4][32] f32
                              const float* __restrict__ pre_b,    // [128] f32
                              const _Float16* __restrict__ w,     // f16 arena
                              _Float16* __restrict__ zout)        // [B*T][128]
{
  // LDS pool: xs f32[96][32] | xh f16[96][32] | hbuf f16[80][256] | skipb f32[80][256]
  __shared__ __align__(16) char smem[141312];
  float*    xs    = (float*)smem;                      // 12288 B
  _Float16* xh    = (_Float16*)(smem + 12288);         //  6144 B
  _Float16* hbuf  = (_Float16*)(smem + 18432);         // 40960 B
  float*    skipb = (float*)(smem + 59392);            // 81920 B
  _Float16* ebuf  = (_Float16*)(smem + 59392);         // reuse skipb: [64][256] f16

  const int b    = blockIdx.y;
  const int t0   = blockIdx.x * TT;
  const int ts   = t0 - 16;                // global t of stored row 0
  const int tid  = threadIdx.x;
  const int wave = tid >> 5;
  const int lane = tid & 31;
  const int half = lane >> 4;
  const int nl   = lane & 15;

  // init: x0 = start_w[c] * input, zero outside [0,T); zero skip accumulator
  for (int i = tid; i < SW * 32; i += 256) {
    int row = i >> 5, c = i & 31;
    int gt = ts + row;
    float v = 0.0f;
    if (gt >= 0 && gt < T_LEN) v = xin[(size_t)b * T_LEN + gt] * start_w[c];
    xs[i] = v;
    xh[i] = (_Float16)v;
  }
  for (int i = tid; i < CW * 256; i += 256) skipb[i] = 0.0f;
  __syncthreads();

  for (int l = 0; l < 4; ++l) {
    const _Float16* fw0 = w + OFF_EF0 + (size_t)l * 256 * 32;
    const _Float16* fw1 = w + OFF_EF1 + (size_t)l * 256 * 32;
    const _Float16* gw0 = w + OFF_EG0 + (size_t)l * 256 * 32;
    const _Float16* gw1 = w + OFF_EG1 + (size_t)l * 256 * 32;
    const _Float16* sw  = w + OFF_ES  + (size_t)l * 256 * 256;
    const _Float16* rw  = w + OFF_ER  + (size_t)l * 32 * 256;
    const float*    sb  = skip_b + l * 256;
    const float*    rb  = res_b  + l * 32;

    // Stage A: h = tanh(filt)*sigmoid(gate). y[t] = W0 x[t-1] + W1 x[t+1].
    for (int nt = wave; nt < 16; nt += 8) {
      const int n0 = nt * 16;
      v16h bF0 = load_b16(fw0, 32, n0, 0);
      v16h bF1 = load_b16(fw1, 32, n0, 0);
      v16h bG0 = load_b16(gw0, 32, n0, 0);
      v16h bG1 = load_b16(gw1, 32, n0, 0);
      const int nn = n0 + nl;
      for (int mt = 0; mt < 5; ++mt) {
        const int mrow = mt * 16;
        v16h aL = load_a_lds(xh, SOFF + mrow - 1, 32, 0);
        v16h aR = load_a_lds(xh, SOFF + mrow + 1, 32, 0);
        v8f accf = wmma16(aL, bF0, vzero());
        accf     = wmma16(aR, bF1, accf);
        v8f accg = wmma16(aL, bG0, vzero());
        accg     = wmma16(aR, bG1, accg);
#pragma unroll
        for (int r = 0; r < 8; ++r) {
          float h = fast_tanh(accf[r]) * fast_sigmoid(accg[r]);
          hbuf[(mrow + r + 8 * half) * 256 + nn] = (_Float16)h;
        }
      }
    }
    __syncthreads();

    // Stage B/C: 18 N-jobs = 16 skip (256->256, +=) + 2 res (256->32, x +=)
    for (int j = wave; j < 18; j += 8) {
      if (j < 16) {
        const int n0 = j * 16;
        v16h bs[8];
#pragma unroll
        for (int ks = 0; ks < 8; ++ks) bs[ks] = load_b16(sw, 256, n0, ks * 32);
        const int nn = n0 + nl;
        const float bias = sb[nn];
        for (int mt = 0; mt < 5; ++mt) {
          const int mrow = mt * 16;
          v8f acc = vzero();
#pragma unroll
          for (int ks = 0; ks < 8; ++ks)
            acc = wmma16(load_a_lds(hbuf, mrow, 256, ks * 32), bs[ks], acc);
#pragma unroll
          for (int r = 0; r < 8; ++r)
            skipb[(mrow + r + 8 * half) * 256 + nn] += acc[r] + bias;
        }
      } else {
        const int n0 = (j - 16) * 16;
        v16h bs[8];
#pragma unroll
        for (int ks = 0; ks < 8; ++ks) bs[ks] = load_b16(rw, 256, n0, ks * 32);
        const int nn = n0 + nl;
        const float bias = rb[nn];
        for (int mt = 0; mt < 5; ++mt) {
          const int mrow = mt * 16;
          v8f acc = vzero();
#pragma unroll
          for (int ks = 0; ks < 8; ++ks)
            acc = wmma16(load_a_lds(hbuf, mrow, 256, ks * 32), bs[ks], acc);
#pragma unroll
          for (int r = 0; r < 8; ++r) {
            int srow = SOFF + mrow + r + 8 * half;
            int gt = ts + srow;
            if (gt >= 0 && gt < T_LEN) {   // keep zero padding outside sequence
              int idx = srow * 32 + nn;
              float nv = xs[idx] + acc[r] + bias;
              xs[idx] = nv;
              xh[idx] = (_Float16)nv;
            }
          }
        }
      }
    }
    __syncthreads();
  }

  // convert skip_tot -> f16 (A operand for the end conv)
  for (int i = tid; i < CW * 256; i += 256) hbuf[i] = (_Float16)skipb[i];
  __syncthreads();

  // end conv (256->256) + tanh -> ebuf (interior rows only; skipb is dead)
  for (int nt = wave; nt < 16; nt += 8) {
    const int n0 = nt * 16;
    v16h bs[8];
#pragma unroll
    for (int ks = 0; ks < 8; ++ks) bs[ks] = load_b16(w + OFF_EEND, 256, n0, ks * 32);
    const int nn = n0 + nl;
    for (int mt = 0; mt < 4; ++mt) {
      v8f acc = vzero();
#pragma unroll
      for (int ks = 0; ks < 8; ++ks)
        acc = wmma16(load_a_lds(hbuf, SOFF + mt * 16, 256, ks * 32), bs[ks], acc);
#pragma unroll
      for (int r = 0; r < 8; ++r)
        ebuf[(mt * 16 + r + 8 * half) * 256 + nn] = (_Float16)fast_tanh(acc[r]);
    }
  }
  __syncthreads();

  // pre conv (256->128) + bias -> z f16 in workspace, layout [pos][128]
  for (int nt = wave; nt < 8; nt += 8) {
    const int n0 = nt * 16;
    v16h bs[8];
#pragma unroll
    for (int ks = 0; ks < 8; ++ks) bs[ks] = load_b16(w + OFF_PRE, 256, n0, ks * 32);
    const int nn = n0 + nl;
    const float bias = pre_b[nn];
    for (int mt = 0; mt < 4; ++mt) {
      v8f acc = vzero();
#pragma unroll
      for (int ks = 0; ks < 8; ++ks)
        acc = wmma16(load_a_lds(ebuf, mt * 16, 256, ks * 32), bs[ks], acc);
#pragma unroll
      for (int r = 0; r < 8; ++r) {
        long p = (long)b * T_LEN + t0 + mt * 16 + r + 8 * half;
        zout[p * 128 + nn] = (_Float16)(acc[r] + bias);
      }
    }
  }
}

// ---------------------------------------------------------------------------
// VQ: scores = ||e||^2 - 2 z.e via WMMA, argmin over 256 codes, gather emb.
// In place: z (f16) is overwritten with quantized (f16). 128 positions/block.
// ---------------------------------------------------------------------------
__launch_bounds__(256)
__global__ void wn_vq_kernel(const float* __restrict__ emb,   // [256][128] f32
                             const _Float16* __restrict__ w,  // f16 arena
                             _Float16* __restrict__ zq)       // [N][128] in/out
{
  __shared__ __align__(16) _Float16 embh[256 * 128];   // 64 KB
  __shared__ float    enorm[256];
  __shared__ float    redv[8 * 16 * 16];
  __shared__ int      redi[8 * 16 * 16];
  __shared__ int      selidx[128];

  const int tid = threadIdx.x, wave = tid >> 5, lane = tid & 31;
  const int half = lane >> 4, nl = lane & 15;

  // stage pre-converted f16 codebook: pure uint4 copies
  {
    const uint4* src = (const uint4*)(w + OFF_EMB);
    uint4* dst = (uint4*)embh;
    for (int i = tid; i < 256 * 128 / 8; i += 256) dst[i] = src[i];
  }
  {
    float s = 0.0f;
    for (int d = 0; d < 128; ++d) { float v = emb[tid * 128 + d]; s += v * v; }
    enorm[tid] = s;
  }
  __syncthreads();

  const long rowbase = (long)blockIdx.x * 128 + wave * 16;

  // A fragments (this wave's 16 z-rows) are invariant across code tiles: hoist.
  v16h az[4];
#pragma unroll
  for (int ks = 0; ks < 4; ++ks) az[ks] = load_a_glb(zq, rowbase, 128, ks * 32);

  float best[8];
  int   bidx[8];
#pragma unroll
  for (int r = 0; r < 8; ++r) { best[r] = 3.4e38f; bidx[r] = 0; }

  for (int nt = 0; nt < 16; ++nt) {
    v8f acc = vzero();
#pragma unroll
    for (int ks = 0; ks < 4; ++ks)
      acc = wmma16(az[ks], load_b16(embh, 128, nt * 16, ks * 32), acc);
    int n = nt * 16 + nl;
    float en = enorm[n];
#pragma unroll
    for (int r = 0; r < 8; ++r) {
      float d = en - 2.0f * acc[r];
      if (d < best[r]) { best[r] = d; bidx[r] = n; }
    }
  }
#pragma unroll
  for (int r = 0; r < 8; ++r) {
    int row = r + 8 * half;
    redv[(wave * 16 + row) * 16 + nl] = best[r];
    redi[(wave * 16 + row) * 16 + nl] = bidx[r];
  }
  __syncthreads();

  if (tid < 128) {
    float bv = 3.4e38f; int bi = 1 << 30;
    for (int c = 0; c < 16; ++c) {
      float v = redv[tid * 16 + c];
      int   i2 = redi[tid * 16 + c];
      if (v < bv || (v == bv && i2 < bi)) { bv = v; bi = i2; }
    }
    selidx[tid] = bi;
  }
  __syncthreads();

  if (tid < 128) {
    long p = (long)blockIdx.x * 128 + tid;
    const uint4* src = (const uint4*)(embh + selidx[tid] * 128);
    uint4* dst = (uint4*)(zq + p * 128);
#pragma unroll
    for (int d = 0; d < 16; ++d) dst[d] = src[d];
  }
}

// ---------------------------------------------------------------------------
// Decoder: 129 channels, pre-padded f16 weights (144 x 160), fused stack
// ---------------------------------------------------------------------------
__launch_bounds__(256)
__global__ void wn_dec_kernel(const _Float16* __restrict__ zq,   // [B*T][128] quantized
                              const float* __restrict__ labels,  // [B]
                              const float* __restrict__ skip_b,  // [4][129] f32
                              const float* __restrict__ res_b,   // [4][32] f32
                              const float* __restrict__ end_w,   // [129] f32
                              const _Float16* __restrict__ w,    // f16 arena
                              float* __restrict__ out)           // [B*T]
{
  // pool: xs f32[96][32] | xh f16[96][32] | { abuf f16[96][160]  OR
  //                                          hbuf f16[80][160] + skipb f32[80][160] }
  __shared__ __align__(16) char smem[95232];
  float*    xs    = (float*)smem;                 // 12288 B
  _Float16* xh    = (_Float16*)(smem + 12288);    //  6144 B
  _Float16* abuf  = (_Float16*)(smem + 18432);    // 30720 B (init/start only)
  _Float16* hbuf  = (_Float16*)(smem + 18432);    // 25600 B
  float*    skipb = (float*)(smem + 44032);       // 51200 B

  const int b    = blockIdx.y;
  const int t0   = blockIdx.x * TT;
  const int ts   = t0 - 16;
  const int tid  = threadIdx.x;
  const int wave = tid >> 5;
  const int lane = tid & 31;
  const int half = lane >> 4;
  const int nl   = lane & 15;

  const float labv = labels[b];
  const unsigned short labh = __builtin_bit_cast(unsigned short, (_Float16)labv);

  // dec_in = [quantized(128) | label(1) | zeros(pad)] via 8-half vector chunks
  for (int i = tid; i < SW * (DKP / 8); i += 256) {
    int row = i / (DKP / 8), c8 = i % (DKP / 8);
    int gt = ts + row;
    uint4 v = {0u, 0u, 0u, 0u};
    if (gt >= 0 && gt < T_LEN) {
      if (c8 < 16) {
        v = *(const uint4*)(zq + ((long)b * T_LEN + gt) * 128 + c8 * 8);
      } else if (c8 == 16) {
        v.x = (unsigned int)labh;   // col 128 = label, cols 129..135 = 0
      }
    }
    *(uint4*)(abuf + row * DKP + c8 * 8) = v;
  }
  __syncthreads();

  // start conv 129->32 (no bias): x0 over all 96 stored rows
  for (int nt = wave; nt < 2; nt += 8) {
    const int n0 = nt * 16;
    v16h bs[5];
#pragma unroll
    for (int ks = 0; ks < 5; ++ks) bs[ks] = load_b16(w + OFF_DST, DKP, n0, ks * 32);
    const int nn = n0 + nl;
    for (int mt = 0; mt < 6; ++mt) {
      v8f acc = vzero();
#pragma unroll
      for (int ks = 0; ks < 5; ++ks)
        acc = wmma16(load_a_lds(abuf, mt * 16, DKP, ks * 32), bs[ks], acc);
#pragma unroll
      for (int r = 0; r < 8; ++r) {
        int idx = (mt * 16 + r + 8 * half) * 32 + nn;
        xs[idx] = acc[r];
        xh[idx] = (_Float16)acc[r];
      }
    }
  }
  __syncthreads();

  // abuf is dead: zero hbuf (incl. K-pad columns) and skip accumulator
  for (int i = tid; i < CW * DKP; i += 256) { hbuf[i] = (_Float16)0.0f; skipb[i] = 0.0f; }
  __syncthreads();

  for (int l = 0; l < 4; ++l) {
    const _Float16* fw0 = w + OFF_DF0 + (size_t)l * DNP * 32;
    const _Float16* fw1 = w + OFF_DF1 + (size_t)l * DNP * 32;
    const _Float16* gw0 = w + OFF_DG0 + (size_t)l * DNP * 32;
    const _Float16* gw1 = w + OFF_DG1 + (size_t)l * DNP * 32;
    const _Float16* sw  = w + OFF_DSK + (size_t)l * DNP * DKP;
    const _Float16* rw  = w + OFF_DR  + (size_t)l * 32 * DKP;
    const float*    sb  = skip_b + l * 129;
    const float*    rb  = res_b  + l * 32;

    // Stage A: filt/gate 32->129 (9 N-jobs; pad rows are pre-zeroed weights)
    for (int nt = wave; nt < 9; nt += 8) {
      const int n0 = nt * 16;
      v16h bF0 = load_b16(fw0, 32, n0, 0);
      v16h bF1 = load_b16(fw1, 32, n0, 0);
      v16h bG0 = load_b16(gw0, 32, n0, 0);
      v16h bG1 = load_b16(gw1, 32, n0, 0);
      const int nn = n0 + nl;
      for (int mt = 0; mt < 5; ++mt) {
        const int mrow = mt * 16;
        v16h aL = load_a_lds(xh, SOFF + mrow - 1, 32, 0);
        v16h aR = load_a_lds(xh, SOFF + mrow + 1, 32, 0);
        v8f accf = wmma16(aL, bF0, vzero());
        accf     = wmma16(aR, bF1, accf);
        v8f accg = wmma16(aL, bG0, vzero());
        accg     = wmma16(aR, bG1, accg);
#pragma unroll
        for (int r = 0; r < 8; ++r) {
          float h = fast_tanh(accf[r]) * fast_sigmoid(accg[r]);  // 0 on pad ch
          hbuf[(mrow + r + 8 * half) * DKP + nn] = (_Float16)h;
        }
      }
    }
    __syncthreads();

    // Stage B/C: 11 N-jobs = 9 skip (129->129, +=) + 2 res (129->32, x +=)
    for (int j = wave; j < 11; j += 8) {
      if (j < 9) {
        const int n0 = j * 16;
        v16h bs[5];
#pragma unroll
        for (int ks = 0; ks < 5; ++ks) bs[ks] = load_b16(sw, DKP, n0, ks * 32);
        const int nn = n0 + nl;
        const float bias = (nn < 129) ? sb[nn] : 0.0f;
        for (int mt = 0; mt < 5; ++mt) {
          const int mrow = mt * 16;
          v8f acc = vzero();
#pragma unroll
          for (int ks = 0; ks < 5; ++ks)
            acc = wmma16(load_a_lds(hbuf, mrow, DKP, ks * 32), bs[ks], acc);
#pragma unroll
          for (int r = 0; r < 8; ++r)
            skipb[(mrow + r + 8 * half) * DKP + nn] += acc[r] + bias;
        }
      } else {
        const int n0 = (j - 9) * 16;
        v16h bs[5];
#pragma unroll
        for (int ks = 0; ks < 5; ++ks) bs[ks] = load_b16(rw, DKP, n0, ks * 32);
        const int nn = n0 + nl;
        const float bias = rb[nn];
        for (int mt = 0; mt < 5; ++mt) {
          const int mrow = mt * 16;
          v8f acc = vzero();
#pragma unroll
          for (int ks = 0; ks < 5; ++ks)
            acc = wmma16(load_a_lds(hbuf, mrow, DKP, ks * 32), bs[ks], acc);
#pragma unroll
          for (int r = 0; r < 8; ++r) {
            int srow = SOFF + mrow + r + 8 * half;
            int gt = ts + srow;
            if (gt >= 0 && gt < T_LEN) {
              int idx = srow * 32 + nn;
              float nv = xs[idx] + acc[r] + bias;
              xs[idx] = nv;
              xh[idx] = (_Float16)nv;
            }
          }
        }
      }
    }
    __syncthreads();
  }

  // end conv 129->1 + tanh(tanh(.)) on interior rows
  if (tid < TT) {
    const float* sp = skipb + (SOFF + tid) * DKP;
    float s = 0.0f;
    for (int c = 0; c < 129; ++c) s += sp[c] * end_w[c];
    out[(size_t)b * T_LEN + t0 + tid] = fast_tanh(fast_tanh(s));
  }
}

// ---------------------------------------------------------------------------
extern "C" void kernel_launch(void* const* d_in, const int* in_sizes, int n_in,
                              void* d_out, int out_size, void* d_ws, size_t ws_size,
                              hipStream_t stream) {
  (void)in_sizes; (void)n_in; (void)out_size; (void)ws_size;

  const float* x            = (const float*)d_in[0];
  const float* labels       = (const float*)d_in[1];
  const float* enc_start_w  = (const float*)d_in[2];
  const float* enc_filter_w = (const float*)d_in[3];
  const float* enc_gate_w   = (const float*)d_in[4];
  const float* enc_skip_w   = (const float*)d_in[5];
  const float* enc_skip_b   = (const float*)d_in[6];
  const float* enc_res_w    = (const float*)d_in[7];
  const float* enc_res_b    = (const float*)d_in[8];
  const float* enc_end_w    = (const float*)d_in[9];
  const float* pre_w        = (const float*)d_in[10];
  const float* pre_b        = (const float*)d_in[11];
  const float* vq_emb       = (const float*)d_in[12];
  const float* dec_start_w  = (const float*)d_in[13];
  const float* dec_filter_w = (const float*)d_in[14];
  const float* dec_gate_w   = (const float*)d_in[15];
  const float* dec_skip_w   = (const float*)d_in[16];
  const float* dec_skip_b   = (const float*)d_in[17];
  const float* dec_res_w    = (const float*)d_in[18];
  const float* dec_res_b    = (const float*)d_in[19];
  const float* dec_end_w    = (const float*)d_in[20];

  _Float16* zq = (_Float16*)d_ws;                        // [B*T][128] f16
  _Float16* w  = zq + (size_t)B_SZ * T_LEN * 128;        // f16 weight arena

  dim3 blk(256);
  wn_pack_kernel<<<dim3(512), blk, 0, stream>>>(
      enc_filter_w, enc_gate_w, enc_skip_w, enc_res_w, enc_end_w, pre_w,
      vq_emb, dec_start_w, dec_filter_w, dec_gate_w, dec_skip_w, dec_res_w, w);

  wn_enc_kernel<<<dim3(T_LEN / TT, B_SZ), blk, 0, stream>>>(
      x, enc_start_w, enc_skip_b, enc_res_b, pre_b, w, zq);

  wn_vq_kernel<<<dim3((B_SZ * T_LEN) / 128), blk, 0, stream>>>(vq_emb, w, zq);

  wn_dec_kernel<<<dim3(T_LEN / TT, B_SZ), blk, 0, stream>>>(
      zq, labels, dec_skip_b, dec_res_b, dec_end_w, w, (float*)d_out);
}
